// GraphNet_83107617177654
// MI455X (gfx1250) — compile-verified, matrix operand order checked by reading, not run on hardware
//
#include <hip/hip_runtime.h>
#include <hip/hip_bf16.h>

// ---------------- WMMA fragment helpers (gfx1250, wave32) ----------------

typedef _Float16 v16h __attribute__((ext_vector_type(16)));
typedef float    v8f  __attribute__((ext_vector_type(8)));

union Frag16 {
    v16h v;
    _Float16 h[16];
    unsigned int u[8];
};
union AccF {
    v8f v;
    float f[8];
};

__device__ __forceinline__ v8f zero8() {
    v8f z = {0.f, 0.f, 0.f, 0.f, 0.f, 0.f, 0.f, 0.f};
    return z;
}

// K index held by VGPR v for 16-bit A/B fragments of v_wmma_*_16x16x32_*:
// lanes 0-15 (half=0): v0..3 -> K 0..7, v4..7 -> K 16..23
// lanes 16-31 (half=1): v0..3 -> K 8..15, v4..7 -> K 24..31
__device__ __forceinline__ int kmap(int half, int v) {
    return (v < 4) ? (half * 8 + v * 2) : (16 + half * 8 + (v - 4) * 2);
}

__device__ __forceinline__ v8f wmma32(v16h a, v16h b, v8f c) {
    // (neg_a, A, neg_b, B, c_mod, C, reuse_a, reuse_b)
    return __builtin_amdgcn_wmma_f32_16x16x32_f16(false, a, false, b, (short)0, c,
                                                  false, false);
}

// A fragment from a row-major f16 tile in LDS: a[m * stride + k]
__device__ __forceinline__ v16h load_a_rm(const _Float16* a, int stride, int kbase,
                                          int lane) {
    Frag16 f;
    int m = lane & 15, half = lane >> 4;
#pragma unroll
    for (int v = 0; v < 8; ++v) {
        int k = kbase + kmap(half, v);
        f.u[v] = *(const unsigned int*)(a + m * stride + k);  // (k even -> aligned)
    }
    return f.v;
}

// B fragment from an N-major (transposed) f16 weight tile in LDS: wt[n * kstride + k]
__device__ __forceinline__ v16h load_b_t(const _Float16* wt, int kstride, int ntile,
                                         int kbase, int lane) {
    Frag16 f;
    int n = ntile * 16 + (lane & 15);
    int half = lane >> 4;
#pragma unroll
    for (int v = 0; v < 8; ++v) {
        int k = kbase + kmap(half, v);
        f.u[v] = *(const unsigned int*)(wt + n * kstride + k);
    }
    return f.v;
}

// C/D layout: VGPR v, lane l -> row = v + 8*(l>>4), col = l&15
__device__ __forceinline__ void bias_relu(AccF& c, const float* __restrict__ bias,
                                          int ncol0, int lane, int nvalid) {
    int nl = lane & 15;
    float bv = (nl < nvalid) ? bias[ncol0 + nl] : 0.f;
#pragma unroll
    for (int v = 0; v < 8; ++v) {
        float t = c.f[v] + bv;
        c.f[v] = t > 0.f ? t : 0.f;
    }
}

__device__ __forceinline__ void store_tile_f16(_Float16* dst, int stride, int ncol0,
                                               const AccF& c, int lane) {
    int col = ncol0 + (lane & 15), half = lane >> 4;
#pragma unroll
    for (int v = 0; v < 8; ++v)
        dst[(v + 8 * half) * stride + col] = (_Float16)c.f[v];
}

__device__ __forceinline__ void wait_lds() {
    asm volatile("s_wait_dscnt 0" ::: "memory");
}

// ---------------- problem constants ----------------
#define NCONST 188
#define PFEAT 16
#define HIDDEN 128
#define DE 5
#define DO_ 6
#define NTGT 5
#define BATCH 32
#define NEDGE_PER_R 187  // senders per receiver (contiguous edge group)

// =========================================================================
// Kernel 1: edge MLP (fr: 32 -> 128 -> 64 -> 5) fused with receiver-sum.
// One block per (batch, receiver). 128 threads = 4 wave32; each wave owns
// M-tiles t, t+4, t+8 of the 192-row (187 valid) edge group.
// =========================================================================
__global__ __launch_bounds__(128) void edge_mlp_kernel(
    const float* __restrict__ x,
    const float* __restrict__ w1, const float* __restrict__ b1,
    const float* __restrict__ w2, const float* __restrict__ b2,
    const float* __restrict__ w3, const float* __restrict__ b3,
    float* __restrict__ ebar_g) {
    __shared__ _Float16 xh[NCONST * PFEAT];       // x[b] as f16          6016 B
    __shared__ _Float16 w1t[HIDDEN * 32];         // [col][k]             8 KB
    __shared__ _Float16 w2t[64 * HIDDEN];         // [col][k]            16 KB
    __shared__ _Float16 w3t[16 * 64];             // [col][k], cols>=5=0  2 KB
    __shared__ _Float16 h1s[4][16 * HIDDEN];      // per-wave scratch    16 KB
    __shared__ _Float16 h2s[4][16 * 64];          // per-wave scratch     8 KB
    __shared__ float    esums[4][16];

    const int tid = threadIdx.x;
    const int b = blockIdx.x / NCONST;  // batch
    const int r = blockIdx.x % NCONST;  // receiver node

    // -------- stage inputs / weights into LDS (f16) --------
    for (int i = tid; i < NCONST * PFEAT; i += 128)
        xh[i] = (_Float16)x[b * NCONST * PFEAT + i];
    for (int i = tid; i < HIDDEN * 32; i += 128) {
        int col = i >> 5, k = i & 31;
        w1t[i] = (_Float16)w1[k * HIDDEN + col];
    }
    for (int i = tid; i < 64 * HIDDEN; i += 128) {
        int col = i >> 7, k = i & 127;
        w2t[i] = (_Float16)w2[k * 64 + col];
    }
    for (int i = tid; i < 16 * 64; i += 128) {
        int col = i >> 6, k = i & 63;
        w3t[i] = (col < DE) ? (_Float16)w3[k * DE + col] : (_Float16)0.f;
    }
    __syncthreads();

    const int wave = tid >> 5, lane = tid & 31;
    const int m = lane & 15, half = lane >> 4;
    _Float16* h1w = h1s[wave];
    _Float16* h2w = h2s[wave];

    float esum = 0.f;  // per-lane partial of Ebar column (lane&15)

    for (int t = wave; t < 12; t += 4) {
        // ---- build layer-1 A fragment straight from xh ----
        // edge e -> row: cols 0..15 = x[r], cols 16..31 = x[s], s = e + (e>=r)
        int e = t * 16 + m;
        int ec = e > (NEDGE_PER_R - 1) ? (NEDGE_PER_R - 1) : e;  // pad rows: dup last
        int s = ec + (ec >= r ? 1 : 0);
        Frag16 a1;
#pragma unroll
        for (int v = 0; v < 8; ++v) {
            int k = kmap(half, v);  // even; dword pair never crosses the 16 boundary
            const _Float16* src = (k < 16) ? (xh + r * PFEAT + k)
                                           : (xh + s * PFEAT + (k - 16));
            a1.u[v] = *(const unsigned int*)src;
        }

        // ---- layer 1: K=32 (one WMMA), N=128 (8 tiles) ----
#pragma unroll
        for (int n = 0; n < 8; ++n) {
            AccF c;
            c.v = wmma32(a1.v, load_b_t(w1t, 32, n, 0, lane), zero8());
            bias_relu(c, b1, n * 16, lane, 16);
            store_tile_f16(h1w, HIDDEN, n * 16, c, lane);
        }
        wait_lds();

        // ---- layer 2: K=128 (4 steps), N=64 (4 tiles) ----
#pragma unroll
        for (int n = 0; n < 4; ++n) {
            AccF c;
            c.v = zero8();
#pragma unroll
            for (int kk = 0; kk < 4; ++kk)
                c.v = wmma32(load_a_rm(h1w, HIDDEN, kk * 32, lane),
                             load_b_t(w2t, HIDDEN, n, kk * 32, lane), c.v);
            bias_relu(c, b2, n * 16, lane, 16);
            store_tile_f16(h2w, 64, n * 16, c, lane);
        }
        wait_lds();

        // ---- layer 3: K=64 (2 steps), N=16 (5 valid) ----
        AccF c3;
        c3.v = zero8();
#pragma unroll
        for (int kk = 0; kk < 2; ++kk)
            c3.v = wmma32(load_a_rm(h2w, 64, kk * 32, lane),
                          load_b_t(w3t, 64, 0, kk * 32, lane), c3.v);
        bias_relu(c3, b3, 0, lane, DE);

        // ---- accumulate Ebar columns over valid rows ----
#pragma unroll
        for (int v = 0; v < 8; ++v) {
            int row = t * 16 + v + 8 * half;
            if (row < NEDGE_PER_R) esum += c3.f[v];
        }
    }

    // lanes c and c+16 hold the two row-halves of column c: fold them.
    esum += __shfl_xor(esum, 16);
    if (lane < 16) esums[wave][lane] = esum;
    __syncthreads();
    if (tid < DE) {
        float tot = esums[0][tid] + esums[1][tid] + esums[2][tid] + esums[3][tid];
        ebar_g[(b * NCONST + r) * DE + tid] = tot;  // Ebar[b][n][De]
    }
}

// =========================================================================
// Kernel 2: node MLP (fo: 21(pad 32) -> 128 -> 64 -> 6) over 6016 rows.
// 376 M-tiles; 4 waves per block, one tile per wave; grid = 94.
// =========================================================================
__device__ __forceinline__ float node_feat(const float* __restrict__ x,
                                           const float* __restrict__ eb,
                                           int row, int col) {
    if (col < PFEAT) return x[row * PFEAT + col];
    if (col < PFEAT + DE) return eb[row * DE + (col - PFEAT)];
    return 0.f;
}

__global__ __launch_bounds__(128) void node_mlp_kernel(
    const float* __restrict__ x, const float* __restrict__ ebar_g,
    const float* __restrict__ w1, const float* __restrict__ b1,
    const float* __restrict__ w2, const float* __restrict__ b2,
    const float* __restrict__ w3, const float* __restrict__ b3,
    float* __restrict__ o_g) {
    __shared__ _Float16 w1t[HIDDEN * 32];   // [col][k], k>=21 zero
    __shared__ _Float16 w2t[64 * HIDDEN];
    __shared__ _Float16 w3t[16 * 64];       // cols >= 6 zero
    __shared__ _Float16 h1s[4][16 * HIDDEN];
    __shared__ _Float16 h2s[4][16 * 64];

    const int tid = threadIdx.x;
    for (int i = tid; i < HIDDEN * 32; i += 128) {
        int col = i >> 5, k = i & 31;
        w1t[i] = (k < PFEAT + DE) ? (_Float16)w1[k * HIDDEN + col] : (_Float16)0.f;
    }
    for (int i = tid; i < 64 * HIDDEN; i += 128) {
        int col = i >> 7, k = i & 127;
        w2t[i] = (_Float16)w2[k * 64 + col];
    }
    for (int i = tid; i < 16 * 64; i += 128) {
        int col = i >> 6, k = i & 63;
        w3t[i] = (col < DO_) ? (_Float16)w3[k * DO_ + col] : (_Float16)0.f;
    }
    __syncthreads();

    const int wave = tid >> 5, lane = tid & 31;
    const int m = lane & 15, half = lane >> 4;
    const int tile = blockIdx.x * 4 + wave;  // 6016 rows = 376 tiles exactly
    const int row = tile * 16 + m;
    _Float16* h1w = h1s[wave];
    _Float16* h2w = h2s[wave];

    // A fragment for layer 1 built from global x + Ebar (f32 -> f16 pack)
    Frag16 a1;
#pragma unroll
    for (int v = 0; v < 8; ++v) {
        int k = kmap(half, v);
        a1.h[2 * v]     = (_Float16)node_feat(x, ebar_g, row, k);
        a1.h[2 * v + 1] = (_Float16)node_feat(x, ebar_g, row, k + 1);
    }

#pragma unroll
    for (int n = 0; n < 8; ++n) {
        AccF c;
        c.v = wmma32(a1.v, load_b_t(w1t, 32, n, 0, lane), zero8());
        bias_relu(c, b1, n * 16, lane, 16);
        store_tile_f16(h1w, HIDDEN, n * 16, c, lane);
    }
    wait_lds();

#pragma unroll
    for (int n = 0; n < 4; ++n) {
        AccF c;
        c.v = zero8();
#pragma unroll
        for (int kk = 0; kk < 4; ++kk)
            c.v = wmma32(load_a_rm(h1w, HIDDEN, kk * 32, lane),
                         load_b_t(w2t, HIDDEN, n, kk * 32, lane), c.v);
        bias_relu(c, b2, n * 16, lane, 16);
        store_tile_f16(h2w, 64, n * 16, c, lane);
    }
    wait_lds();

    AccF c3;
    c3.v = zero8();
#pragma unroll
    for (int kk = 0; kk < 2; ++kk)
        c3.v = wmma32(load_a_rm(h2w, 64, kk * 32, lane),
                      load_b_t(w3t, 64, 0, kk * 32, lane), c3.v);
    bias_relu(c3, b3, 0, lane, DO_);

    // O[b][n][Do] flat = [B, 1128] for the classifier
    int col = lane & 15;
    if (col < DO_) {
#pragma unroll
        for (int v = 0; v < 8; ++v) {
            int rr = tile * 16 + v + 8 * half;
            o_g[rr * DO_ + col] = c3.f[v];
        }
    }
}

// =========================================================================
// Kernel 3: tiny classifier layers in plain f32 (4.6 MMAC total, negligible)
// =========================================================================
__global__ void fc_layer_kernel(const float* __restrict__ in,
                                const float* __restrict__ w,
                                const float* __restrict__ bias,
                                float* __restrict__ out,
                                int rows, int fin, int fout, int do_relu) {
    int idx = blockIdx.x * blockDim.x + threadIdx.x;
    if (idx >= rows * fout) return;
    int r = idx / fout, c = idx % fout;
    float acc = bias[c];
    for (int k = 0; k < fin; ++k) acc += in[r * fin + k] * w[k * fout + c];
    out[idx] = do_relu ? fmaxf(acc, 0.f) : acc;
}

// ---------------- host-side launcher ----------------
extern "C" void kernel_launch(void* const* d_in, const int* in_sizes, int n_in,
                              void* d_out, int out_size, void* d_ws, size_t ws_size,
                              hipStream_t stream) {
    (void)in_sizes; (void)n_in; (void)out_size; (void)ws_size;
    const float* x     = (const float*)d_in[0];
    const float* fr1w  = (const float*)d_in[1];
    const float* fr1b  = (const float*)d_in[2];
    const float* fr2w  = (const float*)d_in[3];
    const float* fr2b  = (const float*)d_in[4];
    const float* fr3w  = (const float*)d_in[5];
    const float* fr3b  = (const float*)d_in[6];
    const float* fo1w  = (const float*)d_in[7];
    const float* fo1b  = (const float*)d_in[8];
    const float* fo2w  = (const float*)d_in[9];
    const float* fo2b  = (const float*)d_in[10];
    const float* fo3w  = (const float*)d_in[11];
    const float* fo3b  = (const float*)d_in[12];
    const float* fc1w  = (const float*)d_in[13];
    const float* fc1b  = (const float*)d_in[14];
    const float* fc2w  = (const float*)d_in[15];
    const float* fc2b  = (const float*)d_in[16];
    const float* fc3w  = (const float*)d_in[17];
    const float* fc3b  = (const float*)d_in[18];

    float* ws   = (float*)d_ws;
    float* Ebar = ws;                               // [32*188*5]
    float* O    = Ebar + BATCH * NCONST * DE;       // [32*188*6] == [32,1128]
    float* H1   = O + BATCH * NCONST * DO_;         // [32*128]
    float* H2   = H1 + BATCH * HIDDEN;              // [32*64]

    edge_mlp_kernel<<<BATCH * NCONST, 128, 0, stream>>>(
        x, fr1w, fr1b, fr2w, fr2b, fr3w, fr3b, Ebar);

    node_mlp_kernel<<<(BATCH * NCONST) / (16 * 4), 128, 0, stream>>>(
        x, Ebar, fo1w, fo1b, fo2w, fo2b, fo3w, fo3b, O);

    fc_layer_kernel<<<(BATCH * HIDDEN + 255) / 256, 256, 0, stream>>>(
        O, fc1w, fc1b, H1, BATCH, DO_ * NCONST, HIDDEN, 1);
    fc_layer_kernel<<<(BATCH * 64 + 255) / 256, 256, 0, stream>>>(
        H1, fc2w, fc2b, H2, BATCH, HIDDEN, 64, 1);
    fc_layer_kernel<<<1, 256, 0, stream>>>(
        H2, fc3w, fc3b, (float*)d_out, BATCH, 64, NTGT, 0);
}
